// LPKTPlusV1_11493332484345
// MI455X (gfx1250) — compile-verified, matrix operand order checked by reading, not run on hardware
//
#include <hip/hip_runtime.h>
#include <math.h>

// ---------------------------------------------------------------------------
// LPKT+ forward on MI455X (gfx1250, wave32).
// One workgroup (256 threads = 8 waves) per batch element. Recurrent state
// h[b] (256x128 f32) lives in LDS for the whole scan; the per-step
// [256x128] @ [128x128]^T GEMM runs on v_wmma_f32_16x16x32_bf16 (bf16 in,
// f32 accumulate), fused with the sigmoid gate + state update epilogue.
// ---------------------------------------------------------------------------

#define BSZ   32
#define SLEN  128
#define CDIM  256
#define KDIM  128

#define H_STRIDE  132   // f32/row (528 B: 16B-aligned rows, bank-spread)
#define W4_STRIDE 136   // bf16/row (272 B: 16B-aligned rows, bank-spread)

typedef __attribute__((ext_vector_type(16))) __bf16 v16bf;
typedef __attribute__((ext_vector_type(8)))  float  v8f;

// one v_exp_f32 + one v_rcp_f32 (no IEEE div chain)
__device__ __forceinline__ float sigmoidf_fast(float x) {
    return __builtin_amdgcn_rcpf(1.0f + __expf(-x));
}

__global__ __launch_bounds__(256)
void lpkt_scan_kernel(const int*   __restrict__ qseq,     // [B,S]
                      const int*   __restrict__ itseq,    // [B,S]
                      const int*   __restrict__ utseq,    // [B,S]
                      const float* __restrict__ correct,  // [B,S]
                      const float* __restrict__ E_q,      // [NQ+1,128]
                      const float* __restrict__ E_it,     // [NIT+1,128]
                      const float* __restrict__ E_ut,     // [NUT+1,128]
                      const float* __restrict__ qmat,     // [NQ+1,256]
                      const float* __restrict__ h0,       // [256,128]
                      const float* __restrict__ W1,       // [128,384]
                      const float* __restrict__ b1,       // [128]
                      const float* __restrict__ W2,       // [128,512]
                      const float* __restrict__ b2,       // [128]
                      const float* __restrict__ W3,       // [128,512]
                      const float* __restrict__ b3,       // [128]
                      const float* __restrict__ W4,       // [128,384] = [W4h|W4l|W4i]
                      const float* __restrict__ b4,       // [128]
                      const float* __restrict__ W5,       // [128,256]
                      const float* __restrict__ b5,       // [128]
                      float*       __restrict__ pred)     // [B,S]
{
    const int b    = blockIdx.x;
    const int tid  = threadIdx.x;      // 0..255
    const int wave = tid >> 5;         // 0..7
    const int lane = tid & 31;         // wave32
    const int jt   = tid & 127;        // output-row index for split phases
    const int half = tid >> 7;         // 0/1 : which half of a split dot

    extern __shared__ char smem_raw[];
    float*  sh_h      = (float*)smem_raw;            // 256*132
    float*  sh_cat    = sh_h      + CDIM * H_STRIDE; // 512
    float*  sh_x      = sh_cat    + 512;             // 256 (q_emb|ut_emb, step i)
    float*  sh_al     = sh_x      + 256;             // 128 all_learning[:,i]
    float*  sh_alprev = sh_al     + 128;             // 128 all_learning[:,i-1]
    float*  sh_it     = sh_alprev + 128;             // 128 it_emb[:,i]
    float*  sh_LG     = sh_it     + 128;             // 128
    float*  sh_bias   = sh_LG     + 128;             // 128
    float*  sh_ht     = sh_bias   + 128;             // 128 h_tilde
    float*  sh_qembn  = sh_ht     + 128;             // 128 q_emb[:,i+1]
    float*  sh_red    = sh_qembn  + 128;             // 128 (wave partials)
    float*  sh_qet    = sh_red    + 128;             // 256 qe[:,i]
    float*  sh_qen    = sh_qet    + CDIM;            // 256 qe[:,i+1]
    float*  sh_pa     = sh_qen    + CDIM;            // 256 split-dot scratch A
    float*  sh_pb     = sh_pa     + 256;             // 256 split-dot scratch B
    __bf16* sh_w4     = (__bf16*)(sh_pb + 256);      // 128*136 bf16 W4h

    // -------- init: h = h0, W4h -> bf16 LDS, qe_t = qmat[qseq[b,0]] --------
    for (int idx = tid; idx < CDIM * KDIM; idx += 256) {
        int r = idx >> 7, c = idx & 127;
        sh_h[r * H_STRIDE + c] = h0[idx];
    }
    for (int idx = tid; idx < KDIM * KDIM; idx += 256) {
        int r = idx >> 7, c = idx & 127;
        sh_w4[r * W4_STRIDE + c] = (__bf16)W4[r * 384 + c];   // W4h = W4[:, :128]
    }
    {
        int q0 = qseq[b * SLEN];
        if (tid < CDIM) sh_qet[tid] = qmat[q0 * CDIM + tid];
    }
    if (tid < 128) sh_alprev[tid] = 0.0f;   // learning0 = 0
    if (tid == 0)  pred[b * SLEN] = 0.0f;   // pred[:,0] = 0

    // step-invariant per-thread registers
    float rb1 = 0.f, rb2 = 0.f, rb3 = 0.f, rb4 = 0.f, rb5 = 0.f, csum = 0.f;
    if (tid < KDIM) {
        rb1 = b1[tid]; rb2 = b2[tid]; rb3 = b3[tid]; rb4 = b4[tid]; rb5 = b5[tid];
        const float* w = W1 + tid * 384;
        for (int k = 256; k < 384; ++k) csum += w[k];   // fold c_emb (corr bcast)
    }
    __syncthreads();

    // h_tilde0 = sum_c qe_t[c] * h0[c, j]
    if (tid < KDIM) {
        float s = 0.0f;
        for (int c = 0; c < CDIM; ++c) s += sh_qet[c] * sh_h[c * H_STRIDE + tid];
        sh_ht[tid] = s;
    }
    __syncthreads();

    const int m     = lane & 15;            // row/col within 16-tile
    const int khalf = (lane >> 4) << 3;     // 0 or 8 : k-offset of lane half

    // ============================ scan over steps ==========================
    for (int i = 0; i < SLEN - 1; ++i) {
        const int   qi   = qseq [b * SLEN + i];
        const int   qn   = qseq [b * SLEN + i + 1];
        const int   iti  = itseq[b * SLEN + i];
        const int   uti  = utseq[b * SLEN + i];
        const float corr = correct[b * SLEN + i];

        // warm next step's embedding rows through GL2 (global_prefetch_b8)
        if (tid == 0 && i + 1 < SLEN - 1) {
            __builtin_prefetch(&E_it[itseq[b * SLEN + i + 1] * KDIM], 0, 0);
            __builtin_prefetch(&E_ut[utseq[b * SLEN + i + 1] * KDIM], 0, 0);
        }

        // ---- gathers ----
        if (tid < KDIM) {
            sh_it[tid]       = E_it[iti * KDIM + tid];
            sh_qembn[tid]    = E_q [qn  * KDIM + tid];
            sh_x[tid]        = E_q [qi  * KDIM + tid];   // q_emb[:,i]
            sh_x[KDIM + tid] = E_ut[uti * KDIM + tid];   // ut_emb[:,i]
        }
        if (tid < CDIM) sh_qen[tid] = qmat[qn * CDIM + tid];
        __syncthreads();

        // ---- all_learning[:,i] + cat build + roll learning_pre ----
        if (tid < KDIM) {
            const float* w = W1 + tid * 384;
            float acc = rb1 + corr * csum;
            for (int k = 0; k < 256; ++k) acc += sh_x[k] * w[k];
            sh_al[tid] = acc;
            // cat = [learning_pre, it, learning, h_tilde_pre]
            sh_cat[tid]            = sh_alprev[tid];
            sh_cat[KDIM + tid]     = sh_it[tid];
            sh_cat[2 * KDIM + tid] = acc;
            sh_cat[3 * KDIM + tid] = sh_ht[tid];
            sh_alprev[tid]         = acc;   // thread-private roll
        }
        __syncthreads();

        // ---- gate dots: 512-long, split across 256 threads ----
        {
            const int   k0 = half * 256;
            const float* w2 = W2 + jt * 512 + k0;
            const float* w3 = W3 + jt * 512 + k0;
            float a2 = 0.f, a3 = 0.f;
            for (int k = 0; k < 256; ++k) {
                float cv = sh_cat[k0 + k];
                a2 += cv * w2[k];
                a3 += cv * w3[k];
            }
            sh_pa[tid] = a2;
            sh_pb[tid] = a3;
        }
        __syncthreads();
        // LG = sigmoid(a3) * (tanh(a2)+1)/2 == sigmoid(a3) * sigmoid(2*a2)
        if (tid < KDIM) {
            float a2 = rb2 + sh_pa[tid] + sh_pa[tid + 128];
            float a3 = rb3 + sh_pb[tid] + sh_pb[tid + 128];
            sh_LG[tid] = sigmoidf_fast(a3) * sigmoidf_fast(2.0f * a2);
        }
        __syncthreads();

        // ---- bias_row = LG@W4l^T + it@W4i^T + b4 ----
        if (tid < KDIM) {
            const float* w = W4 + tid * 384;
            float acc = rb4;
            for (int k = 0; k < KDIM; ++k)
                acc += sh_LG[k] * w[128 + k] + sh_it[k] * w[256 + k];
            sh_bias[tid] = acc;
        }
        __syncthreads();

        // ---- WMMA phase: gamma_f pre-act = h_pre @ W4h^T, fused update ----
        // wave exclusively owns h rows [wave*32, wave*32+32): no intra-phase
        // barrier; A fragments cached before any write to those rows.
        {
            v16bf afrag[2][4];
            #pragma unroll
            for (int rt = 0; rt < 2; ++rt) {
                const float* hrow = sh_h + (wave * 32 + rt * 16 + m) * H_STRIDE;
                #pragma unroll
                for (int kf = 0; kf < 4; ++kf) {
                    #pragma unroll
                    for (int p = 0; p < 8; ++p) {
                        int k = kf * 32 + ((p >> 2) << 4) + khalf + ((p & 3) << 1);
                        afrag[rt][kf][2 * p]     = (__bf16)hrow[k];
                        afrag[rt][kf][2 * p + 1] = (__bf16)hrow[k + 1];
                    }
                }
            }
            for (int ct = 0; ct < 8; ++ct) {
                const int colBase = ct * 16;
                const __bf16* wrow = sh_w4 + (colBase + m) * W4_STRIDE;
                v16bf bfrag[4];              // B[k][n] = W4h[n][k]
                #pragma unroll
                for (int kf = 0; kf < 4; ++kf) {
                    #pragma unroll
                    for (int p = 0; p < 8; ++p) {
                        int k = kf * 32 + ((p >> 2) << 4) + khalf + ((p & 3) << 1);
                        bfrag[kf][2 * p]     = wrow[k];
                        bfrag[kf][2 * p + 1] = wrow[k + 1];
                    }
                }
                const int   j     = colBase + m;     // N = lane & 15
                const float biasj = sh_bias[j];
                const float lgj   = sh_LG[j];
                #pragma unroll
                for (int rt = 0; rt < 2; ++rt) {
                    v8f acc = {};
                    #pragma unroll
                    for (int kf = 0; kf < 4; ++kf)
                        acc = __builtin_amdgcn_wmma_f32_16x16x32_bf16(
                            false, afrag[rt][kf], false, bfrag[kf],
                            (short)0, acc, false, false);
                    const int rowBase = wave * 32 + rt * 16;
                    #pragma unroll
                    for (int r = 0; r < 8; ++r) {
                        int c = rowBase + r + khalf;   // M = r + 8*(lane>=16)
                        float gf   = sigmoidf_fast(acc[r] + biasj);
                        float* hp  = &sh_h[c * H_STRIDE + j];
                        *hp = sh_qet[c] * lgj + gf * (*hp);
                    }
                }
            }
        }
        __syncthreads();

        // ---- h_tilde = sum_c qe_n[c]*h[c,:]  (split over 256 threads) ----
        {
            const int c0 = half * 128;
            float s = 0.0f;
            for (int c = 0; c < 128; ++c)
                s += sh_qen[c0 + c] * sh_h[(c0 + c) * H_STRIDE + jt];
            sh_pb[tid] = s;
            sh_qet[tid] = sh_qen[tid];      // roll qe_t (read only in WMMA phase)
        }
        __syncthreads();
        if (tid < KDIM) sh_ht[tid] = sh_pb[tid] + sh_pb[tid + 128];
        __syncthreads();

        // ---- y = mean(sigmoid([qemb_n, h_tilde] @ W5^T + b5)) ----
        {
            const float* w5 = W5 + jt * 256 + half * 128;
            const float* v  = half ? sh_ht : sh_qembn;
            float p = 0.0f;
            for (int k = 0; k < KDIM; ++k) p += v[k] * w5[k];
            sh_pa[tid] = p;
        }
        __syncthreads();
        if (tid < KDIM) {
            float z = sigmoidf_fast(rb5 + sh_pa[tid] + sh_pa[tid + 128]);
            #pragma unroll
            for (int off = 16; off >= 1; off >>= 1)
                z += __shfl_down(z, off, 32);
            if (lane == 0) sh_red[wave] = z;   // waves 0..3 active
        }
        __syncthreads();
        if (tid == 0)
            pred[b * SLEN + i + 1] =
                (sh_red[0] + sh_red[1] + sh_red[2] + sh_red[3]) * (1.0f / (float)KDIM);
        __syncthreads();
    }
}

extern "C" void kernel_launch(void* const* d_in, const int* in_sizes, int n_in,
                              void* d_out, int out_size, void* d_ws, size_t ws_size,
                              hipStream_t stream) {
    (void)in_sizes; (void)n_in; (void)out_size; (void)d_ws; (void)ws_size;
    const int*   qseq    = (const int*)  d_in[0];
    const int*   itseq   = (const int*)  d_in[1];
    const int*   utseq   = (const int*)  d_in[2];
    const float* correct = (const float*)d_in[3];
    const float* E_q     = (const float*)d_in[4];
    const float* E_it    = (const float*)d_in[5];
    const float* E_ut    = (const float*)d_in[6];
    const float* qmat    = (const float*)d_in[7];
    const float* h0      = (const float*)d_in[8];
    const float* W1      = (const float*)d_in[9];
    const float* b1      = (const float*)d_in[10];
    const float* W2      = (const float*)d_in[11];
    const float* b2      = (const float*)d_in[12];
    const float* W3      = (const float*)d_in[13];
    const float* b3      = (const float*)d_in[14];
    const float* W4      = (const float*)d_in[15];
    const float* b4      = (const float*)d_in[16];
    const float* W5      = (const float*)d_in[17];
    const float* b5      = (const float*)d_in[18];
    float*       pred    = (float*)d_out;

    const size_t shmem =
        (size_t)(CDIM * H_STRIDE      // h state
                 + 512 + 256          // cat, x
                 + 8 * 128            // al, alprev, it, LG, bias, ht, qembn, red
                 + 2 * CDIM           // qe_t, qe_n
                 + 2 * 256) * 4       // split-dot scratch pa, pb
        + (size_t)(KDIM * W4_STRIDE) * 2;   // W4h bf16

    lpkt_scan_kernel<<<dim3(BSZ), dim3(256), shmem, stream>>>(
        qseq, itseq, utseq, correct, E_q, E_it, E_ut, qmat, h0,
        W1, b1, W2, b2, W3, b3, W4, b4, W5, b5, pred);
}
